// AttentionPooling_15281493639508
// MI455X (gfx1250) — matrix operand order repaired and checked by reading, not compiled, not run on hardware
//
#include <hip/hip_runtime.h>
#include <hip/hip_bf16.h>

// ---------------------------------------------------------------------------
// AttentionPooling for MI455X (gfx1250, wave32, WMMA bf16 16x16x32, f32 acc)
//
//   att = sigmoid(cat(input64, final128) @ lin_w^T + lin_b)     [N,128]
//   g   = att * (final128 @ last_w^T + last_b)                  [N,128]
//   out = segment_sum(g, graph_index (sorted), 50000)           [50000,128]
//
// fp32 accuracy via bf16 hi/lo split: x ~= hi + lo, x*w ~= hi*wh + hi*wl + lo*wh
// ---------------------------------------------------------------------------

typedef __bf16 bf16_t;
typedef __attribute__((ext_vector_type(16))) __bf16 v16bf;
typedef __attribute__((ext_vector_type(8)))  __bf16 v8bf;
typedef __attribute__((ext_vector_type(4)))  __bf16 v4bf;
typedef __attribute__((ext_vector_type(8)))  float  v8f;
typedef __attribute__((ext_vector_type(4)))  float  v4f;

#define MOL_C   64
#define HID_C   128
#define CAT_C   192
#define ROW_STRIDE 200          // bf16 elems per LDS row (192 + 8 pad, 400B, 16B aligned)
#define WAVES_PER_BLOCK 4

union V16U { v16bf v; v8bf h[2]; };

__device__ __forceinline__ void split_bf16(float x, bf16_t& h, bf16_t& l) {
    h = (bf16_t)x;                 // RNE truncation to bf16
    l = (bf16_t)(x - (float)h);    // residual, exact subtraction
}

__device__ __forceinline__ v16bf wmma_bf16_fma(v16bf a, v16bf b, v8f& c) {
    c = __builtin_amdgcn_wmma_f32_16x16x32_bf16(false, a, false, b, (short)0, c,
                                                false, false);
    return a;
}

// A-fragment (16x32 bf16, M x K) from LDS, row-major with ROW_STRIDE.
// lane layout: M = lane&15; half = lane>>4; elems 0..15 = K kbase+half*16+0..15
// (contiguous, matching the B fragment).
__device__ __forceinline__ v16bf load_afrag(const bf16_t* A, int m, int kbase, int half) {
    V16U a;
    const bf16_t* p = A + m * ROW_STRIDE + kbase + half * 16;
    a.h[0] = *(const v8bf*)(p);
    a.h[1] = *(const v8bf*)(p + 8);
    return a.v;
}

// B-fragment (32x16 bf16, K x N). W stored row-major [n][k] (== B column-major),
// lane holds column N = lane&15, K = kbase + half*16 .. +15 (contiguous 32B).
__device__ __forceinline__ v16bf load_bfrag(const bf16_t* __restrict__ W, int kdim,
                                            int n, int kbase, int half) {
    return *(const v16bf*)(W + (size_t)n * kdim + kbase + half * 16);
}

__device__ __forceinline__ void atomic_add_f32(float* p, float v) {
    __hip_atomic_fetch_add(p, v, __ATOMIC_RELAXED, __HIP_MEMORY_SCOPE_AGENT);
}

// ---------------------------------------------------------------------------
// Prep: split fp32 weights into bf16 hi/lo planes in workspace.
// ---------------------------------------------------------------------------
__global__ void ap_prep_weights(const float* __restrict__ lin_w,
                                const float* __restrict__ last_w,
                                bf16_t* __restrict__ lin_hi, bf16_t* __restrict__ lin_lo,
                                bf16_t* __restrict__ last_hi, bf16_t* __restrict__ last_lo) {
    int i = blockIdx.x * blockDim.x + threadIdx.x;
    if (i < HID_C * CAT_C) {
        bf16_t h, l;
        split_bf16(lin_w[i], h, l);
        lin_hi[i] = h; lin_lo[i] = l;
    } else {
        int j = i - HID_C * CAT_C;
        if (j < HID_C * HID_C) {
            bf16_t h, l;
            split_bf16(last_w[j], h, l);
            last_hi[j] = h; last_lo[j] = l;
        }
    }
}

// ---------------------------------------------------------------------------
// Main: one wave per 16-node tile.
// ---------------------------------------------------------------------------
__global__ __launch_bounds__(32 * WAVES_PER_BLOCK)
void ap_main_kernel(const float* __restrict__ input_rep,
                    const float* __restrict__ final_rep,
                    const int*   __restrict__ graph_index,
                    const float* __restrict__ lin_b,
                    const float* __restrict__ last_b,
                    const bf16_t* __restrict__ lin_hi,  const bf16_t* __restrict__ lin_lo,
                    const bf16_t* __restrict__ last_hi, const bf16_t* __restrict__ last_lo,
                    float* __restrict__ out,
                    int numTiles) {
    __shared__ bf16_t lds_hi[WAVES_PER_BLOCK][16 * ROW_STRIDE];
    __shared__ bf16_t lds_lo[WAVES_PER_BLOCK][16 * ROW_STRIDE];

    const int lane = threadIdx.x & 31;
    const int wave = threadIdx.x >> 5;
    const int tile = blockIdx.x * WAVES_PER_BLOCK + wave;
    const bool valid = (tile < numTiles);
    const int rowBase = tile * 16;

    bf16_t* Ahi = lds_hi[wave];
    bf16_t* Alo = lds_lo[wave];

    if (valid) {
        // Stage input_rep tile: 16 rows x 64 f32 -> bf16 hi/lo at LDS cols [0,64)
        for (int u = lane; u < 16 * (MOL_C / 4); u += 32) {
            int r = u / (MOL_C / 4);
            int c = (u % (MOL_C / 4)) * 4;
            v4f x = *(const v4f*)(input_rep + (size_t)(rowBase + r) * MOL_C + c);
            v4bf h, l;
            #pragma unroll
            for (int k = 0; k < 4; ++k) { bf16_t hh, ll; split_bf16(x[k], hh, ll); h[k] = hh; l[k] = ll; }
            *(v4bf*)(Ahi + r * ROW_STRIDE + c) = h;
            *(v4bf*)(Alo + r * ROW_STRIDE + c) = l;
        }
        // Stage final_rep tile: 16 rows x 128 f32 -> bf16 hi/lo at LDS cols [64,192)
        for (int u = lane; u < 16 * (HID_C / 4); u += 32) {
            int r = u / (HID_C / 4);
            int c = (u % (HID_C / 4)) * 4;
            v4f x = *(const v4f*)(final_rep + (size_t)(rowBase + r) * HID_C + c);
            v4bf h, l;
            #pragma unroll
            for (int k = 0; k < 4; ++k) { bf16_t hh, ll; split_bf16(x[k], hh, ll); h[k] = hh; l[k] = ll; }
            *(v4bf*)(Ahi + r * ROW_STRIDE + MOL_C + c) = h;
            *(v4bf*)(Alo + r * ROW_STRIDE + MOL_C + c) = l;
        }
    }
    __syncthreads();   // uniform: all waves reach this (also orders LDS cross-lane)
    if (!valid) return;

    const int half = lane >> 4;      // 0: rows 0-7 / K-low ; 1: rows 8-15 / K-high
    const int n16  = lane & 15;      // output column within 16-wide N tile

    // Sorted graph indices for this lane's 8 output rows (rows half*8 + r)
    int gidx[8];
    #pragma unroll
    for (int r = 0; r < 8; ++r) gidx[r] = graph_index[rowBase + half * 8 + r];

    #pragma unroll
    for (int nt = 0; nt < HID_C / 16; ++nt) {
        const int ncol = nt * 16 + n16;

        // ---- att_pre = cat @ lin_w^T + lin_b : accumulate over K = 192 ----
        float b0 = lin_b[ncol];
        v8f acc;
        #pragma unroll
        for (int r = 0; r < 8; ++r) acc[r] = b0;

        #pragma unroll
        for (int kt = 0; kt < CAT_C / 32; ++kt) {
            v16bf ah = load_afrag(Ahi, n16, kt * 32, half);
            v16bf al = load_afrag(Alo, n16, kt * 32, half);
            v16bf bh = load_bfrag(lin_hi, CAT_C, ncol, kt * 32, half);
            v16bf bl = load_bfrag(lin_lo, CAT_C, ncol, kt * 32, half);
            acc = __builtin_amdgcn_wmma_f32_16x16x32_bf16(false, ah, false, bh, (short)0, acc, false, false);
            acc = __builtin_amdgcn_wmma_f32_16x16x32_bf16(false, ah, false, bl, (short)0, acc, false, false);
            acc = __builtin_amdgcn_wmma_f32_16x16x32_bf16(false, al, false, bh, (short)0, acc, false, false);
        }

        // ---- att = sigmoid(att_pre) ----
        v8f att;
        #pragma unroll
        for (int r = 0; r < 8; ++r) att[r] = 1.0f / (1.0f + __expf(-acc[r]));

        // ---- h = final @ last_w^T + last_b : accumulate over K = 128 ----
        float b1 = last_b[ncol];
        v8f acc2;
        #pragma unroll
        for (int r = 0; r < 8; ++r) acc2[r] = b1;

        #pragma unroll
        for (int kt = 0; kt < HID_C / 32; ++kt) {
            v16bf ah = load_afrag(Ahi, n16, MOL_C + kt * 32, half);
            v16bf al = load_afrag(Alo, n16, MOL_C + kt * 32, half);
            v16bf bh = load_bfrag(last_hi, HID_C, ncol, kt * 32, half);
            v16bf bl = load_bfrag(last_lo, HID_C, ncol, kt * 32, half);
            acc2 = __builtin_amdgcn_wmma_f32_16x16x32_bf16(false, ah, false, bh, (short)0, acc2, false, false);
            acc2 = __builtin_amdgcn_wmma_f32_16x16x32_bf16(false, ah, false, bl, (short)0, acc2, false, false);
            acc2 = __builtin_amdgcn_wmma_f32_16x16x32_bf16(false, al, false, bh, (short)0, acc2, false, false);
        }

        // ---- gate + run-length segment sum (indices sorted) + atomic flush ----
        float s  = att[0] * acc2[0];
        int  cur = gidx[0];
        #pragma unroll
        for (int r = 1; r < 8; ++r) {
            float gv = att[r] * acc2[r];
            if (gidx[r] == cur) {
                s += gv;
            } else {
                atomic_add_f32(out + (size_t)cur * HID_C + ncol, s);
                cur = gidx[r];
                s = gv;
            }
        }
        atomic_add_f32(out + (size_t)cur * HID_C + ncol, s);
    }
}

// ---------------------------------------------------------------------------
// Launch
// ---------------------------------------------------------------------------
extern "C" void kernel_launch(void* const* d_in, const int* in_sizes, int n_in,
                              void* d_out, int out_size, void* d_ws, size_t ws_size,
                              hipStream_t stream) {
    const float* input_rep   = (const float*)d_in[0];
    const float* final_rep   = (const float*)d_in[1];
    const int*   graph_index = (const int*)d_in[2];
    const float* lin_w       = (const float*)d_in[3];
    const float* lin_b       = (const float*)d_in[4];
    const float* last_w      = (const float*)d_in[5];
    const float* last_b      = (const float*)d_in[6];
    float* out = (float*)d_out;

    const int nNodes = in_sizes[2];          // graph_index length == N_NODES

    // Workspace layout: bf16 hi/lo planes of both weight matrices (~160 KB)
    bf16_t* lin_hi  = (bf16_t*)d_ws;
    bf16_t* lin_lo  = lin_hi  + HID_C * CAT_C;
    bf16_t* last_hi = lin_lo  + HID_C * CAT_C;
    bf16_t* last_lo = last_hi + HID_C * HID_C;

    // Output is accumulated via atomics -> must start zeroed every call.
    hipMemsetAsync(d_out, 0, (size_t)out_size * sizeof(float), stream);

    int prepThreads = HID_C * CAT_C + HID_C * HID_C;
    ap_prep_weights<<<(prepThreads + 255) / 256, 256, 0, stream>>>(
        lin_w, last_w, lin_hi, lin_lo, last_hi, last_lo);

    int numTiles = nNodes / 16;              // N_NODES = 1e6 -> 62500 tiles
    int blocks   = (numTiles + WAVES_PER_BLOCK - 1) / WAVES_PER_BLOCK;
    ap_main_kernel<<<blocks, 32 * WAVES_PER_BLOCK, 0, stream>>>(
        input_rep, final_rep, graph_index, lin_b, last_b,
        lin_hi, lin_lo, last_hi, last_lo, out, numTiles);
}
